// DenseEdgeConv_63909113364825
// MI455X (gfx1250) — matrix operand order repaired and checked by reading, not compiled, not run on hardware
//
#include <hip/hip_runtime.h>
#include <hip/hip_bf16.h>
#include <float.h>

typedef __attribute__((ext_vector_type(16))) _Float16 v16h;
typedef __attribute__((ext_vector_type(8)))  _Float16 v8h;
typedef __attribute__((ext_vector_type(2)))  _Float16 v2h;
typedef __attribute__((ext_vector_type(8)))  float    v8f;

#define BB 4
#define NN 4096
#define DD 64
#define KK 16
#define FW 320     // D + L*G
#define WAVES 4

// ---------------- Phase 1: brute-force KNN (top-16, excluding self) ----------------
__global__ __launch_bounds__(256) void knn_kernel(const float* __restrict__ pos,
                                                  int* __restrict__ knn)
{
    __shared__ float sx[256], sy[256], sz[256];
    const int tid = threadIdx.x;
    const int bi  = blockIdx.x >> 4;               // 16 blocks per batch (N/256)
    const int ni  = ((blockIdx.x & 15) << 8) + tid;
    const float* pb = pos + (size_t)bi * NN * 3;
    const float qx = pb[ni*3+0], qy = pb[ni*3+1], qz = pb[ni*3+2];

    float bd[KK];
    int   bx[KK];
#pragma unroll
    for (int s = 0; s < KK; ++s) { bd[s] = FLT_MAX; bx[s] = 0; }

    for (int t = 0; t < NN; t += 256) {
        __syncthreads();
        sx[tid] = pb[(t+tid)*3+0];
        sy[tid] = pb[(t+tid)*3+1];
        sz[tid] = pb[(t+tid)*3+2];
        __syncthreads();
#pragma unroll 4
        for (int j = 0; j < 256; ++j) {
            const int jj = t + j;
            const float dx = sx[j]-qx, dy = sy[j]-qy, dz = sz[j]-qz;
            const float d2 = dx*dx + dy*dy + dz*dz;
            if (jj != ni && d2 < bd[KK-1]) {
                bd[KK-1] = d2; bx[KK-1] = jj;
#pragma unroll
                for (int s = KK-1; s > 0; --s) {
                    if (bd[s] < bd[s-1]) {
                        float td = bd[s]; bd[s] = bd[s-1]; bd[s-1] = td;
                        int   ti = bx[s]; bx[s] = bx[s-1]; bx[s-1] = ti;
                    }
                }
            }
        }
    }
    int* rowp = knn + ((size_t)bi*NN + ni) * KK;
#pragma unroll
    for (int s = 0; s < KK; ++s) rowp[s] = bx[s];
}

// ---------------- Phase 2: dense edge-MLP chain via WMMA + max over K ----------------
// One wave per query point: the 16 neighbors are the M=16 rows of every WMMA tile.
// Weights live in LDS *pre-fragmented* into the per-lane WMMA B layout, so a
// B fragment is one contiguous 32B-per-lane load (2x ds_load_b128, no moves).
// Concatenated K space across layers: 768 rows -> 24 chunks of 32.
//   layer0: chunks 0..5   layer1: 6..9   layer2: 10..15   layer3: 16..23
__global__ __launch_bounds__(32*WAVES) void edge_mlp_kernel(
    const float* __restrict__ x, const int* __restrict__ knn,
    const float* __restrict__ Wf,  const float* __restrict__ bf,
    const float* __restrict__ Wm1, const float* __restrict__ bm1,
    const float* __restrict__ Wm2, const float* __restrict__ bm2,
    const float* __restrict__ Wl,  const float* __restrict__ bl,
    float* __restrict__ out)
{
    // wfrag[gc][t][lane][e] : fragment for global chunk gc, n-tile t
    __shared__ __align__(32) _Float16 wfrag[24*4*32*16];   // 96 KB
    __shared__ float bias[4*64];
    __shared__ __align__(16) _Float16 feat[WAVES][16][FW]; // 40 KB

    const int tid = threadIdx.x;
    // cooperative load: shuffle f32 weights into pre-fragmented f16 B layout
    for (int i = tid; i < 24*4*32*16; i += blockDim.x) {
        const int e  = i & 15;
        const int l  = (i >> 4) & 31;
        const int t  = (i >> 9) & 3;
        const int gc = i >> 11;
        const int kglob = (gc << 5) + ((e >> 3) << 4) + ((l >> 4) << 3) + (e & 7);  // 0..767
        const int col   = (t << 4) + (l & 15);
        float w;
        if      (kglob < 192) w = Wf [(kglob      ) * 64 + col];
        else if (kglob < 320) w = Wm1[(kglob - 192) * 64 + col];
        else if (kglob < 512) w = Wm2[(kglob - 320) * 64 + col];
        else                  w = Wl [(kglob - 512) * 64 + col];
        wfrag[i] = (_Float16)w;
    }
    for (int i = tid; i < 64; i += blockDim.x) {
        bias[i] = bf[i]; bias[64+i] = bm1[i]; bias[128+i] = bm2[i]; bias[192+i] = bl[i];
    }
    __syncthreads();

    const int wave = tid >> 5;
    const int lane = tid & 31;
    const int row  = lane & 15;      // M row / output column within tile
    const int hi   = lane >> 4;      // which half of the wave
    _Float16 (*F)[FW] = feat[wave];

    const int khalf = hi << 3;       // A-frag K sub-offset (ISA 16-bit A layout)

    const int nPts   = BB * NN;
    const int stride = gridDim.x * WAVES;
    for (int p = blockIdx.x * WAVES + wave; p < nPts; p += stride) {
        const int b = p >> 12;
        // ---- build edge features: [x_c | x_nb | x_nb - x_c] in F[.][0..191],
        //      plus carried copy of x_c in F[.][256..319].
        //      Each lane owns the contiguous feature pair {2*lane, 2*lane+1}.
        const float* xc = x + (size_t)p * DD;
        const float2 c2 = *(const float2*)(xc + 2*lane);
        const v2h hc = { (_Float16)c2.x, (_Float16)c2.y };
#pragma unroll
        for (int m = 0; m < KK; ++m) {
            *(v2h*)&F[m][      2*lane] = hc;
            *(v2h*)&F[m][256 + 2*lane] = hc;
        }
        const int* nrow = knn + (size_t)p * KK;
        for (int m = 0; m < KK; ++m) {
            const int j = nrow[m];
            const float* xn = x + ((size_t)b * NN + j) * DD;   // L2-resident gather
            const float2 v2 = *(const float2*)(xn + 2*lane);
            const v2h hn = { (_Float16)v2.x,        (_Float16)v2.y };
            const v2h hd = { (_Float16)(v2.x-c2.x), (_Float16)(v2.y-c2.y) };
            *(v2h*)&F[m][ 64 + 2*lane] = hn;
            *(v2h*)&F[m][128 + 2*lane] = hd;
        }

        // layer tables: input base / #chunks / global-chunk base / output base
        const int inbase[4] = {0,   192, 128, 64};
        const int nchk[4]   = {6,   4,   6,   8};
        const int gbase[4]  = {0,   6,   10,  16};
        const int outb[4]   = {192, 128, 64,  0};

#pragma unroll
        for (int L = 0; L < 4; ++L) {
#pragma unroll
            for (int t = 0; t < 4; ++t) {               // 4 tiles of 16 output cols
                const int col = (t << 4) + row;
                v8f acc = {};
#pragma unroll
                for (int kc = 0; kc < nchk[L]; ++kc) {
                    // A fragment: row = edge, 16 halves {khalf..+7, 16+khalf..+7}
                    const int kA = inbase[L] + (kc << 5);
                    const v8h alo = *(const v8h*)&F[row][kA + khalf];
                    const v8h ahi = *(const v8h*)&F[row][kA + 16 + khalf];
                    const v16h a = __builtin_shufflevector(alo, ahi,
                                     0,1,2,3,4,5,6,7,8,9,10,11,12,13,14,15);
                    // B fragment: contiguous 32B per lane from pre-fragmented LDS
                    const int fi = ((((gbase[L] + kc) << 2) + t) << 5) + lane;
                    const v16h bwt = *(const v16h*)&wfrag[fi << 4];
                    acc = __builtin_amdgcn_wmma_f32_16x16x32_f16(
                              false, a, false, bwt, (short)0, acc, false, false);
                }
                const float bv = bias[(L << 6) + col];
                if (L < 3) {
                    // relu + store f16 into this layer's output slot
#pragma unroll
                    for (int r = 0; r < 8; ++r) {
                        float v = acc[r] + bv;
                        v = v > 0.f ? v : 0.f;
                        F[(hi<<3) + r][outb[L] + col] = (_Float16)v;
                    }
                } else {
                    // last layer: no relu; max over the 16 edge rows directly
                    float mx = -FLT_MAX;
#pragma unroll
                    for (int r = 0; r < 8; ++r) {
                        float v = acc[r] + bv;
                        mx = v > mx ? v : mx;
                    }
                    float o = __shfl_xor(mx, 16, 32);   // combine M=0..7 with M=8..15
                    mx = o > mx ? o : mx;
                    if (hi == 0) out[(size_t)p * FW + col] = mx;
                }
            }
        }

        // max over neighbors for carried features [64..319]:
        // each lane owns a contiguous run of 8 halves -> one ds_load_b128 per row
        {
            const int fbase = 64 + lane * 8;
            float mx[8];
#pragma unroll
            for (int e = 0; e < 8; ++e) mx[e] = -FLT_MAX;
#pragma unroll
            for (int m = 0; m < KK; ++m) {
                const v8h r8 = *(const v8h*)&F[m][fbase];
#pragma unroll
                for (int e = 0; e < 8; ++e) {
                    const float v = (float)r8[e];
                    mx[e] = v > mx[e] ? v : mx[e];
                }
            }
            float4 o0 = { mx[0], mx[1], mx[2], mx[3] };
            float4 o1 = { mx[4], mx[5], mx[6], mx[7] };
            *(float4*)&out[(size_t)p * FW + fbase]     = o0;
            *(float4*)&out[(size_t)p * FW + fbase + 4] = o1;
        }
    }
}

extern "C" void kernel_launch(void* const* d_in, const int* in_sizes, int n_in,
                              void* d_out, int out_size, void* d_ws, size_t ws_size,
                              hipStream_t stream) {
    const float* x   = (const float*)d_in[0];
    const float* pos = (const float*)d_in[1];
    const float* Wf  = (const float*)d_in[2];
    const float* bf  = (const float*)d_in[3];
    const float* Wm1 = (const float*)d_in[4];
    const float* bm1 = (const float*)d_in[5];
    const float* Wm2 = (const float*)d_in[6];
    const float* bm2 = (const float*)d_in[7];
    const float* Wl  = (const float*)d_in[8];
    const float* bl  = (const float*)d_in[9];
    float* out = (float*)d_out;
    int*   knn = (int*)d_ws;                 // B*N*K ints = 1 MB scratch

    knn_kernel<<<(BB*NN)/256, 256, 0, stream>>>(pos, knn);
    edge_mlp_kernel<<<256, 32*WAVES, 0, stream>>>(x, knn, Wf, bf, Wm1, bm1,
                                                  Wm2, bm2, Wl, bl, out);
}
